// CrossAttentionLayer_420906795771
// MI455X (gfx1250) — matrix operand order repaired and checked by reading, not compiled, not run on hardware
//
#include <hip/hip_runtime.h>

#define D_DIM 128
#define H_DIM 128
#define NL 256
#define NR 2048
#define EPSF 1e-8f

typedef __attribute__((ext_vector_type(16))) _Float16 v16h;
typedef __attribute__((ext_vector_type(8)))  _Float16 v8h;
typedef __attribute__((ext_vector_type(8)))  float    v8f;

union V16H { v16h v; v8h h[2]; _Float16 e[16]; };

// fast silu: x * rcp(1+exp(-x)) -> v_mul, v_exp, v_add, v_rcp, v_mul
__device__ __forceinline__ float silu(float x) {
    return x * __builtin_amdgcn_rcpf(1.0f + __expf(-x));
}
// fast tanh: (e^{2x}-1) * rcp(e^{2x}+1)
__device__ __forceinline__ float fast_tanh(float x) {
    float e = __expf(2.0f * x);
    return (e - 1.0f) * __builtin_amdgcn_rcpf(e + 1.0f);
}

// ---------------------------------------------------------------------------
// Kernel 1: A_l = h_l @ We1[0:128,:],  A_r = h_r @ We1[128:256,:] + be1
// ---------------------------------------------------------------------------
__global__ void precompute_A_kernel(const float* __restrict__ h_l,
                                    const float* __restrict__ h_r,
                                    const float* __restrict__ We1,
                                    const float* __restrict__ be1,
                                    float* __restrict__ Al,
                                    float* __restrict__ Ar) {
    __shared__ float sh[128];
    const int row = blockIdx.x;
    const int n   = threadIdx.x;
    const bool isL = (row < NL);
    const float* hp = isL ? (h_l + row * D_DIM) : (h_r + (row - NL) * D_DIM);
    sh[n] = hp[n];
    __syncthreads();
    const float* W = We1 + (isL ? 0 : D_DIM * H_DIM);
    float acc = isL ? 0.0f : be1[n];
    #pragma unroll 8
    for (int k = 0; k < D_DIM; ++k) acc += sh[k] * W[k * H_DIM + n];
    if (isL) Al[row * H_DIM + n] = acc;
    else     Ar[(row - NL) * H_DIM + n] = acc;
}

// ---------------------------------------------------------------------------
// Kernel 2: transpose + f32->f16 convert We2, Wc1 into [n][k] layout
// ---------------------------------------------------------------------------
__global__ void convert_weights_kernel(const float* __restrict__ We2,
                                       const float* __restrict__ Wc1,
                                       _Float16* __restrict__ We2T,
                                       _Float16* __restrict__ Wc1T) {
    int idx = blockIdx.x * 256 + threadIdx.x;   // 16384 elements
    int n = idx & 127;
    int k = idx >> 7;
    We2T[n * 128 + k] = (_Float16)We2[k * 128 + n];
    Wc1T[n * 128 + k] = (_Float16)Wc1[k * 128 + n];
}

// ---------------------------------------------------------------------------
// Kernel 3: fused edge pipeline.  grid = (NL/8, NR/128); block = 8 waves.
// ---------------------------------------------------------------------------
__launch_bounds__(256)
__global__ void egnn_edge_kernel(const float* __restrict__ Al,
                                 const float* __restrict__ Ar,
                                 const float* __restrict__ w256,   // We1 row 256
                                 const _Float16* __restrict__ We2T,
                                 const _Float16* __restrict__ Wc1T,
                                 const float* __restrict__ be2,
                                 const float* __restrict__ bc1,
                                 const float* __restrict__ Wc2,
                                 const float* __restrict__ x_l,
                                 const float* __restrict__ x_r,
                                 float* __restrict__ aggWS,
                                 float* __restrict__ transWS) {
    // 136-half row stride: rows 16B-aligned, no 256B-stride bank aliasing
    __shared__ __attribute__((aligned(16))) _Float16 sWe2T[128][136];
    __shared__ __attribute__((aligned(16))) _Float16 sWc1T[128][136];
    __shared__ __attribute__((aligned(16))) _Float16 sM[8][16][136];
    __shared__ float sAl[8][128];
    __shared__ float sW256[128];
    __shared__ float sXr[128][3];
    __shared__ float sWbuf[8][16];

    const int tid  = threadIdx.x;
    const int wave = tid >> 5;
    const int lane = tid & 31;
    const int g    = lane >> 4;
    const int mrow = lane & 15;

    const int iBase = blockIdx.x * 8;
    const int jBase = blockIdx.y * 128;

    for (int idx = tid; idx < 128 * 128; idx += 256) {
        int n = idx >> 7, k = idx & 127;
        sWe2T[n][k] = We2T[idx];
        sWc1T[n][k] = Wc1T[idx];
    }
    for (int idx = tid; idx < 8 * 128; idx += 256)
        sAl[idx >> 7][idx & 127] = Al[(iBase + (idx >> 7)) * H_DIM + (idx & 127)];
    if (tid < 128) sW256[tid] = w256[tid];
    for (int idx = tid; idx < 128 * 3; idx += 256)
        sXr[idx / 3][idx % 3] = x_r[(jBase + idx / 3) * 3 + (idx % 3)];
    __syncthreads();

    const int i = iBase + wave;
    const float xl0 = x_l[i * 3 + 0];
    const float xl1 = x_l[i * 3 + 1];
    const float xl2 = x_l[i * 3 + 2];

    // hoist per-lane bias / Wc2 values (depend only on nt, mrow)
    float be2v[8], bc1v[8], wc2v[8];
    #pragma unroll
    for (int nt = 0; nt < 8; ++nt) {
        const int n = nt * 16 + mrow;
        be2v[nt] = be2[n];
        bc1v[nt] = bc1[n];
        wc2v[nt] = Wc2[n];
    }

    float aggAcc[8];
    #pragma unroll
    for (int nt = 0; nt < 8; ++nt) aggAcc[nt] = 0.0f;
    float tAcc0 = 0.0f, tAcc1 = 0.0f, tAcc2 = 0.0f;

    for (int t = 0; t < 8; ++t) {
        const int jloc = t * 16 + mrow;          // local j row within block
        const int j    = jBase + jloc;

        const float d0 = xl0 - sXr[jloc][0];
        const float d1 = xl1 - sXr[jloc][1];
        const float d2 = xl2 - sXr[jloc][2];
        const float dsq = d0 * d0 + d1 * d1 + d2 * d2;

        // ---- build silu(layer1) directly as WMMA A-fragments in registers
        V16H afr[4];
        const float* ArRow = Ar + j * H_DIM;
        #pragma unroll
        for (int kk = 0; kk < 4; ++kk) {
            const int kb = kk * 32 + g * 8;
            #pragma unroll
            for (int e = 0; e < 8; ++e) {
                const int k0 = kb + e;
                const int k1 = kb + 16 + e;
                float p0 = sAl[wave][k0] + ArRow[k0] + dsq * sW256[k0];
                float p1 = sAl[wave][k1] + ArRow[k1] + dsq * sW256[k1];
                afr[kk].e[e]     = (_Float16)silu(p0);
                afr[kk].e[e + 8] = (_Float16)silu(p1);
            }
        }

        // ---- GEMM 1: m = silu(m1 @ We2 + be2)
        #pragma unroll
        for (int nt = 0; nt < 8; ++nt) {
            const int n = nt * 16 + mrow;
            const _Float16* bp = &sWe2T[n][0];
            // batch all 4 K-step B fragments, then chain the WMMAs
            V16H b[4];
            #pragma unroll
            for (int kk = 0; kk < 4; ++kk) {
                b[kk].h[0] = *(const v8h*)(bp + kk * 32 + g * 8);
                b[kk].h[1] = *(const v8h*)(bp + kk * 32 + 16 + g * 8);
            }
            v8f acc = {};
            #pragma unroll
            for (int kk = 0; kk < 4; ++kk)
                acc = __builtin_amdgcn_wmma_f32_16x16x32_f16(
                    false, afr[kk].v, false, b[kk].v, (short)0, acc, false, false);

            const float bias = be2v[nt];
            float colsum = 0.0f;
            #pragma unroll
            for (int v = 0; v < 8; ++v) {
                float mv = silu(acc[v] + bias);
                colsum += mv;
                sM[wave][v + g * 8][n] = (_Float16)mv;
            }
            aggAcc[nt] += colsum;
        }

        // ---- GEMM 2: t2 = silu(m @ Wc1 + bc1); per-lane dot with Wc2
        float wpart[8];
        #pragma unroll
        for (int v = 0; v < 8; ++v) wpart[v] = 0.0f;

        const _Float16* ap = &sM[wave][mrow][0];
        // A fragments of the m tile (shared across nt)
        V16H a[4];
        #pragma unroll
        for (int kk = 0; kk < 4; ++kk) {
            a[kk].h[0] = *(const v8h*)(ap + kk * 32 + g * 8);
            a[kk].h[1] = *(const v8h*)(ap + kk * 32 + 16 + g * 8);
        }
        #pragma unroll
        for (int nt = 0; nt < 8; ++nt) {
            const int n = nt * 16 + mrow;
            const _Float16* bp = &sWc1T[n][0];
            V16H b[4];
            #pragma unroll
            for (int kk = 0; kk < 4; ++kk) {
                b[kk].h[0] = *(const v8h*)(bp + kk * 32 + g * 8);
                b[kk].h[1] = *(const v8h*)(bp + kk * 32 + 16 + g * 8);
            }
            v8f acc = {};
            #pragma unroll
            for (int kk = 0; kk < 4; ++kk)
                acc = __builtin_amdgcn_wmma_f32_16x16x32_f16(
                    false, a[kk].v, false, b[kk].v, (short)0, acc, false, false);

            const float bias = bc1v[nt];
            const float wc   = wc2v[nt];
            #pragma unroll
            for (int v = 0; v < 8; ++v)
                wpart[v] += silu(acc[v] + bias) * wc;
        }

        // reduce over the 16 column-lanes (xor offsets stay inside 16-group)
        #pragma unroll
        for (int v = 0; v < 8; ++v) {
            float s = wpart[v];
            s += __shfl_xor(s, 1);
            s += __shfl_xor(s, 2);
            s += __shfl_xor(s, 4);
            s += __shfl_xor(s, 8);
            wpart[v] = s;
        }
        if (mrow == 0) {
            #pragma unroll
            for (int v = 0; v < 8; ++v)
                sWbuf[wave][g * 8 + v] = fast_tanh(wpart[v]);
        }

        const float wv = sWbuf[wave][mrow];
        if (g == 0) {
            float dist = sqrtf(dsq + EPSF);
            float f = wv * __builtin_amdgcn_rcpf(dist + EPSF);
            tAcc0 += d0 * f;
            tAcc1 += d1 * f;
            tAcc2 += d2 * f;
        }
    }

    // final agg: combine the two lane-groups, atomically add partials
    #pragma unroll
    for (int nt = 0; nt < 8; ++nt) {
        float s = aggAcc[nt] + __shfl_xor(aggAcc[nt], 16);
        if (g == 0) atomicAdd(&aggWS[i * H_DIM + nt * 16 + mrow], s);
    }
    // final trans: reduce lanes 0-15
    tAcc0 += __shfl_xor(tAcc0, 1); tAcc0 += __shfl_xor(tAcc0, 2);
    tAcc0 += __shfl_xor(tAcc0, 4); tAcc0 += __shfl_xor(tAcc0, 8);
    tAcc1 += __shfl_xor(tAcc1, 1); tAcc1 += __shfl_xor(tAcc1, 2);
    tAcc1 += __shfl_xor(tAcc1, 4); tAcc1 += __shfl_xor(tAcc1, 8);
    tAcc2 += __shfl_xor(tAcc2, 1); tAcc2 += __shfl_xor(tAcc2, 2);
    tAcc2 += __shfl_xor(tAcc2, 4); tAcc2 += __shfl_xor(tAcc2, 8);
    if (lane == 0) {
        atomicAdd(&transWS[i * 3 + 0], tAcc0);
        atomicAdd(&transWS[i * 3 + 1], tAcc1);
        atomicAdd(&transWS[i * 3 + 2], tAcc2);
    }
}

// ---------------------------------------------------------------------------
// Kernel 4: node MLP + coordinate output. One block per left node.
// ---------------------------------------------------------------------------
__global__ void finalize_kernel(const float* __restrict__ h_l,
                                const float* __restrict__ x_l,
                                const float* __restrict__ agg,
                                const float* __restrict__ trans,
                                const float* __restrict__ Wn1,
                                const float* __restrict__ bn1,
                                const float* __restrict__ Wn2,
                                const float* __restrict__ bn2,
                                float* __restrict__ out) {
    __shared__ float sin_[256];
    __shared__ float shid[128];
    const int row = blockIdx.x;
    const int n   = threadIdx.x;   // 128 threads
    sin_[n]       = h_l[row * D_DIM + n];
    sin_[128 + n] = agg[row * H_DIM + n];
    __syncthreads();
    float acc = bn1[n];
    #pragma unroll 8
    for (int k = 0; k < 256; ++k) acc += sin_[k] * Wn1[k * H_DIM + n];
    shid[n] = silu(acc);
    __syncthreads();
    float acc2 = bn2[n];
    #pragma unroll 8
    for (int k = 0; k < 128; ++k) acc2 += shid[k] * Wn2[k * D_DIM + n];
    out[row * D_DIM + n] = h_l[row * D_DIM + n] + acc2;
    if (n < 3) out[NL * D_DIM + row * 3 + n] = x_l[row * 3 + n] + trans[row * 3 + n];
}

// ---------------------------------------------------------------------------
extern "C" void kernel_launch(void* const* d_in, const int* in_sizes, int n_in,
                              void* d_out, int out_size, void* d_ws, size_t ws_size,
                              hipStream_t stream) {
    const float* h_l = (const float*)d_in[0];
    const float* x_l = (const float*)d_in[1];
    const float* h_r = (const float*)d_in[2];
    const float* x_r = (const float*)d_in[3];
    const float* We1 = (const float*)d_in[4];
    const float* be1 = (const float*)d_in[5];
    const float* We2 = (const float*)d_in[6];
    const float* be2 = (const float*)d_in[7];
    const float* Wc1 = (const float*)d_in[8];
    const float* bc1 = (const float*)d_in[9];
    const float* Wc2 = (const float*)d_in[10];
    const float* Wn1 = (const float*)d_in[11];
    const float* bn1 = (const float*)d_in[12];
    const float* Wn2 = (const float*)d_in[13];
    const float* bn2 = (const float*)d_in[14];

    char* ws = (char*)d_ws;
    float*    Alp     = (float*)(ws + 0);               // 256*128*4   = 131072
    float*    Arp     = (float*)(ws + 131072);          // 2048*128*4  = 1048576
    _Float16* We2T    = (_Float16*)(ws + 1179648);      // 32768
    _Float16* Wc1T    = (_Float16*)(ws + 1212416);      // 32768
    float*    aggWS   = (float*)(ws + 1245184);         // 131072
    float*    transWS = (float*)(ws + 1376256);         // 3072

    hipMemsetAsync(aggWS, 0, (NL * H_DIM + NL * 3) * sizeof(float), stream);

    precompute_A_kernel<<<NL + NR, 128, 0, stream>>>(h_l, h_r, We1, be1, Alp, Arp);
    convert_weights_kernel<<<64, 256, 0, stream>>>(We2, Wc1, We2T, Wc1T);

    egnn_edge_kernel<<<dim3(NL / 8, NR / 128), 256, 0, stream>>>(
        Alp, Arp, We1 + 256 * H_DIM, We2T, Wc1T, be2, bc1, Wc2,
        x_l, x_r, aggWS, transWS);

    finalize_kernel<<<NL, 128, 0, stream>>>(h_l, x_l, aggWS, transWS,
                                            Wn1, bn1, Wn2, bn2, (float*)d_out);
}
